// RetNetCLM_69767448756624
// MI455X (gfx1250) — matrix operand order repaired and checked by reading, not compiled, not run on hardware
//
#include <hip/hip_runtime.h>
#include <hip/hip_bf16.h>

#define B_ 4
#define S_ 1024
#define L_ 6
#define D_ 1024
#define F_ 4096
#define H_ 8
#define DH_ 128
#define VOCAB_ 32000
#define T_ (B_*S_)

typedef __attribute__((ext_vector_type(16))) __bf16 v16bf;
typedef __attribute__((ext_vector_type(8)))  __bf16 v8bf;
typedef __attribute__((ext_vector_type(8)))  float  v8f;

__device__ __forceinline__ v8f wmma_bf16(v16bf a, v16bf b, v8f c) {
  // (neg_a, A, neg_b, B, c_mod, C, reuse_a, reuse_b)
  return __builtin_amdgcn_wmma_f32_16x16x32_bf16(false, a, false, b, (short)0, c, false, false);
}

// ---------------------------------------------------------------------------
// Embedding gather: X[t][d] = embed[ids[t]][d]
// ---------------------------------------------------------------------------
__global__ void k_embed(const int* __restrict__ ids, const float* __restrict__ emb,
                        float* __restrict__ X) {
  size_t i = (size_t)blockIdx.x * 256 + threadIdx.x;   // i < T_*D_
  size_t t = i >> 10;                                   // D_ = 1024
  int d = (int)(i & 1023);
  X[i] = emb[(size_t)ids[t] * D_ + d];
}

// ---------------------------------------------------------------------------
// fp32 -> bf16 conversion (weights, activations)
// ---------------------------------------------------------------------------
__global__ void k_f32_to_bf16(const float* __restrict__ s, __bf16* __restrict__ d, size_t n) {
  size_t i = ((size_t)blockIdx.x * 256 + threadIdx.x) * 4;
  if (i + 3 < n) {
    d[i + 0] = (__bf16)s[i + 0];
    d[i + 1] = (__bf16)s[i + 1];
    d[i + 2] = (__bf16)s[i + 2];
    d[i + 3] = (__bf16)s[i + 3];
  } else {
    for (int j = 0; j < 4; ++j)
      if (i + j < n) d[i + j] = (__bf16)s[i + j];
  }
}

// ---------------------------------------------------------------------------
// Row LayerNorm over D=1024, fp32 in, bf16 out (block of 256 per row)
// ---------------------------------------------------------------------------
__global__ void k_layernorm(const float* __restrict__ X, const float* __restrict__ sc,
                            const float* __restrict__ bi, __bf16* __restrict__ out) {
  int row = blockIdx.x, tid = threadIdx.x;
  const float* x = X + (size_t)row * D_;
  float sum = 0.f, sq = 0.f;
  for (int i = tid; i < D_; i += 256) { float v = x[i]; sum += v; sq += v * v; }
  __shared__ float s1[256], s2[256];
  s1[tid] = sum; s2[tid] = sq; __syncthreads();
  for (int off = 128; off > 0; off >>= 1) {
    if (tid < off) { s1[tid] += s1[tid + off]; s2[tid] += s2[tid + off]; }
    __syncthreads();
  }
  float mu = s1[0] * (1.f / D_);
  float var = s2[0] * (1.f / D_) - mu * mu;
  float inv = rsqrtf(var + 1e-5f);
  __bf16* o = out + (size_t)row * D_;
  for (int i = tid; i < D_; i += 256)
    o[i] = (__bf16)((x[i] - mu) * inv * sc[i] + bi[i]);
}

// ---------------------------------------------------------------------------
// Tiled bf16 WMMA GEMM: C[M,N] = A[M,K] * B[K,N] (+bias[N]) (+beta[M,N])
// block = 256 threads = 8 waves (2x4); block tile 64(M) x 128(N); K-step 32.
// Each wave owns a 32x32 tile -> 2 A-frags x 2 B-frags -> 4 WMMAs per k-step.
// FLAGS is compile-time so the epilogue is branch-free.
// ---------------------------------------------------------------------------
template <int FLAGS>
__global__ void k_gemm(const __bf16* __restrict__ A, const __bf16* __restrict__ B,
                       const float* __restrict__ bias, const float* __restrict__ beta,
                       float* __restrict__ C,
                       int M, int N, int K, int lda, int ldb, int ldc) {
  __shared__ alignas(16) __bf16 As[64][40];    // +8 pad keeps 16B alignment per row
  __shared__ alignas(16) __bf16 Bs[128][40];   // stored transposed: Bs[n][k]
  int tid = threadIdx.x;
  int lane = tid & 31, w = tid >> 5;
  int wm = w >> 2, wn = w & 3;                 // 2x4 waves over the block tile
  int m0 = blockIdx.y * 64, n0 = blockIdx.x * 128;
  int lo = lane & 15, hi = lane >> 4;

  v8f acc[4];                                   // [mi*2+ni] 32x32 wave tile
  for (int q = 0; q < 4; ++q)
    for (int r = 0; r < 8; ++r) acc[q][r] = 0.f;

  const int ar = (tid * 8) >> 5, ac = (tid * 8) & 31;     // A: 8 contiguous bf16

  for (int k0 = 0; k0 < K; k0 += 32) {
    // stage A (64x32): one 16B load per thread
    {
      v8bf va = *(const v8bf*)(A + (size_t)(m0 + ar) * lda + k0 + ac);
      *(v8bf*)&As[ar][ac] = va;
    }
    // stage B transposed (32k x 128n): coalesced global, scattered LDS b16
#pragma unroll
    for (int j = 0; j < 16; ++j) {
      int idx = j * 256 + tid;
      int kk = idx >> 7, nn = idx & 127;
      Bs[nn][kk] = B[(size_t)(k0 + kk) * ldb + n0 + nn];
    }
    // prefetch next K-panel into cache while we compute this one
    if (k0 + 32 < K) {
      __builtin_prefetch(A + (size_t)(m0 + ar) * lda + (k0 + 32) + ac, 0, 0);
      __builtin_prefetch(B + (size_t)(k0 + 32 + (tid >> 7)) * ldb + n0 + (tid & 127), 0, 0);
    }
    __syncthreads();

    v16bf a0, a1, b0, b1;
    {
      int row = wm * 32 + lo;
      v8bf x0 = *(const v8bf*)&As[row][hi * 8];
      v8bf x1 = *(const v8bf*)&As[row][16 + hi * 8];
      v8bf x2 = *(const v8bf*)&As[row + 16][hi * 8];
      v8bf x3 = *(const v8bf*)&As[row + 16][16 + hi * 8];
#pragma unroll
      for (int i = 0; i < 8; ++i) {
        a0[i] = x0[i]; a0[8 + i] = x1[i];
        a1[i] = x2[i]; a1[8 + i] = x3[i];
      }
    }
    {
      int col = wn * 32 + lo;
      v8bf y0 = *(const v8bf*)&Bs[col][hi * 8];
      v8bf y1 = *(const v8bf*)&Bs[col][16 + hi * 8];
      v8bf y2 = *(const v8bf*)&Bs[col + 16][hi * 8];
      v8bf y3 = *(const v8bf*)&Bs[col + 16][16 + hi * 8];
#pragma unroll
      for (int i = 0; i < 8; ++i) {
        b0[i] = y0[i]; b0[8 + i] = y1[i];
        b1[i] = y2[i]; b1[8 + i] = y3[i];
      }
    }
    acc[0] = wmma_bf16(a0, b0, acc[0]);
    acc[1] = wmma_bf16(a0, b1, acc[1]);
    acc[2] = wmma_bf16(a1, b0, acc[2]);
    acc[3] = wmma_bf16(a1, b1, acc[3]);
    __syncthreads();
  }

  // epilogue: VGPR r -> M = hi*8 + r within subtile ; N = lo
  float bn[2] = {0.f, 0.f};
  if (FLAGS & 1) {
    bn[0] = bias[n0 + wn * 32 + lo];
    bn[1] = bias[n0 + wn * 32 + 16 + lo];
  }
#pragma unroll
  for (int mi = 0; mi < 2; ++mi) {
#pragma unroll
    for (int ni = 0; ni < 2; ++ni) {
      v8f& a = acc[mi * 2 + ni];
      int n = n0 + wn * 32 + ni * 16 + lo;
#pragma unroll
      for (int r = 0; r < 8; ++r) {
        int m = m0 + wm * 32 + mi * 16 + hi * 8 + r;
        float v = a[r];
        if (FLAGS & 1) v += bn[ni];
        size_t o = (size_t)m * ldc + n;
        if (FLAGS & 2) v += beta[o];
        C[o] = v;
      }
    }
  }
}

// ---------------------------------------------------------------------------
// xPos rotary on Q,K (fp32 GEMM outputs) + pack Q/K/V to bf16.
// Layout [T][H][DH] (== [T][D] with d = h*DH+dh). Thread per (t,h,pair i<64).
// ---------------------------------------------------------------------------
__global__ void k_rope_pack(const float* __restrict__ Qf, const float* __restrict__ Kf,
                            const float* __restrict__ Vf,
                            __bf16* __restrict__ Qb, __bf16* __restrict__ Kb,
                            __bf16* __restrict__ Vb) {
  size_t idx = (size_t)blockIdx.x * 256 + threadIdx.x;  // t*H*64 + h*64 + i
  int i = (int)(idx & 63);
  size_t th = idx >> 6;
  int h = (int)(th & (H_ - 1));
  size_t t = th >> 3;
  int s = (int)(t & (S_ - 1));                           // t = b*S + s
  float base = (2.f * i + 0.4f * DH_) / (1.4f * DH_);
  float scp  = __powf(base, (float)s * (1.f / 512.f));   // xPos scale
  float invf = __powf(10000.f, -(float)i * (1.f / 64.f));
  float ang = (float)s * invf;
  float sn = __sinf(ang), cs = __cosf(ang);
  size_t off = t * D_ + (size_t)h * DH_ + 2 * i;
  float q0 = Qf[off], q1 = Qf[off + 1];
  float k0 = Kf[off], k1 = Kf[off + 1];
  float cq = cs * scp, sq = sn * scp;
  float ck = cs / scp, sk = sn / scp;
  Qb[off]     = (__bf16)(q0 * cq - q1 * sq);
  Qb[off + 1] = (__bf16)(q1 * cq + q0 * sq);
  Kb[off]     = (__bf16)(k0 * ck - k1 * sk);
  Kb[off + 1] = (__bf16)(k1 * ck + k0 * sk);
  Vb[off]     = (__bf16)Vf[off];
  Vb[off + 1] = (__bf16)Vf[off + 1];
}

// ---------------------------------------------------------------------------
// Fused retention, block-cooperative version.
// One 256-thread block owns 128 consecutive queries of one (b,h); wave w
// handles the 16-query tile at qblk*128 + w*16. All waves iterate the same
// key blocks (causal-mask zeroes any out-of-range contribution, so barriers
// stay uniform). K tile and transposed-V tile are staged in LDS with
// coalesced b128 (K) and b16 (V) loads; score tile is transposed per wave through LDS.
// Decay g^(s-t) is computed incrementally (2 expf per key block).
// __launch_bounds__(256,1) lifts the VGPR cap (wave32 allows up to 1024) so
// acc[8] + Q fragments + staging addresses stay resident -> no scratch spills.
// ---------------------------------------------------------------------------
__global__ __launch_bounds__(256, 1)
void k_retention(const __bf16* __restrict__ Qb, const __bf16* __restrict__ Kb,
                 const __bf16* __restrict__ Vb, float* __restrict__ Yh) {
  __shared__ alignas(16) __bf16 Ks[32][136];   // [key][dh], row = 272B (16B mult)
  __shared__ alignas(16) __bf16 Vt[128][40];   // [dh][key], transposed
  __shared__ alignas(16) float  Pl[8][16][32]; // per-wave score transpose slab
  int tid = threadIdx.x;
  int lane = tid & 31, w = tid >> 5;
  int lo = lane & 15, hi = lane >> 4;
  int qblk = blockIdx.x & (S_ / 128 - 1);      // 8 query blocks per (b,h)
  int bh   = blockIdx.x >> 3;
  int h = bh & (H_ - 1), b = bh >> 3;

  const float l32 = logf(1.f / 32.f), l512 = logf(1.f / 512.f);
  float gh = 1.f - __expf(l32 + (l512 - l32) * ((float)h / (float)(H_ - 1)));
  float lg = logf(gh);

  int q0 = qblk * 128 + w * 16;                // this wave's absolute query base
  size_t rowQ = ((size_t)b * S_ + q0 + lo) * D_ + (size_t)h * DH_;

  // Q A-fragments for all 4 K-chunks of DH=128, resident across the key loop
  v16bf aq[4];
#pragma unroll
  for (int kc = 0; kc < 4; ++kc) {
    const __bf16* pq = Qb + rowQ + kc * 32;
    v8bf x0 = *(const v8bf*)(pq + hi * 8);
    v8bf x1 = *(const v8bf*)(pq + 16 + hi * 8);
#pragma unroll
    for (int i = 0; i < 8; ++i) { aq[kc][i] = x0[i]; aq[kc][8 + i] = x1[i]; }
  }

  v8f acc[8];
  for (int c = 0; c < 8; ++c)
    for (int r = 0; r < 8; ++r) acc[c][r] = 0.f;

  // staging coordinates (loop-invariant)
  const int krow = (tid * 16) >> 7, kcol = (tid * 16) & 127;   // K tile: 16/thread

  int nkb = (qblk + 1) * 4;                    // uniform across the block
  for (int kb = 0; kb < nkb; ++kb) {
    int t0 = kb * 32;
    // stage K[t0:t0+32][0:128] (two b128 per thread, coalesced)
    {
      const __bf16* src = Kb + ((size_t)b * S_ + t0 + krow) * D_ + (size_t)h * DH_ + kcol;
      *(v8bf*)&Ks[krow][kcol]     = *(const v8bf*)src;
      *(v8bf*)&Ks[krow][kcol + 8] = *(const v8bf*)(src + 8);
    }
    // stage V transposed: Vt[dh][key] (coalesced global reads)
#pragma unroll
    for (int j = 0; j < 16; ++j) {
      int idx = j * 256 + tid;
      int key = idx >> 7, dh = idx & 127;
      Vt[dh][key] = Vb[((size_t)b * S_ + t0 + key) * D_ + (size_t)h * DH_ + dh];
    }
    __syncthreads();

    // scores: P0 = Q * K[t0..t0+15]^T ; P1 = Q * K[t0+16..t0+31]^T
    v8f P0, P1;
#pragma unroll
    for (int r = 0; r < 8; ++r) { P0[r] = 0.f; P1[r] = 0.f; }
#pragma unroll
    for (int kc = 0; kc < 4; ++kc) {
      v16bf bk0, bk1;
      {
        v8bf y0 = *(const v8bf*)&Ks[lo][kc * 32 + hi * 8];
        v8bf y1 = *(const v8bf*)&Ks[lo][kc * 32 + 16 + hi * 8];
        v8bf z0 = *(const v8bf*)&Ks[16 + lo][kc * 32 + hi * 8];
        v8bf z1 = *(const v8bf*)&Ks[16 + lo][kc * 32 + 16 + hi * 8];
#pragma unroll
        for (int i = 0; i < 8; ++i) {
          bk0[i] = y0[i]; bk0[8 + i] = y1[i];
          bk1[i] = z0[i]; bk1[8 + i] = z1[i];
        }
      }
      P0 = wmma_bf16(aq[kc], bk0, P0);
      P1 = wmma_bf16(aq[kc], bk1, P1);
    }

    // incremental decay + causal mask, stash transposed into per-wave slab
    {
      int s0 = q0 + hi * 8;
      int tt0 = t0 + lo, tt1 = t0 + 16 + lo;
      float cur0 = __expf(lg * (float)(s0 - tt0));   // g^(s0-t); -127<=s0-t<=1016 -> safe
      float cur1 = __expf(lg * (float)(s0 - tt1));
#pragma unroll
      for (int r = 0; r < 8; ++r) {
        int s = s0 + r;
        Pl[w][hi * 8 + r][lo]      = (tt0 <= s) ? P0[r] * cur0 : 0.f;
        Pl[w][hi * 8 + r][16 + lo] = (tt1 <= s) ? P1[r] * cur1 : 0.f;
        cur0 *= gh; cur1 *= gh;
      }
    }
    // wave-private LDS transpose: wait for DS stores before cross-lane reads
    asm volatile("s_wait_dscnt 0" ::: "memory");

    // rebuild score tile as bf16 A-fragment (16 queries x 32 keys)
    v16bf ap;
#pragma unroll
    for (int i = 0; i < 8; ++i) {
      ap[i]     = (__bf16)Pl[w][lo][hi * 8 + i];
      ap[8 + i] = (__bf16)Pl[w][lo][16 + hi * 8 + i];
    }

    // out[16,128] += P[16,32] * V[32,128]  (8 N-chunks of 16; Vt is [dh][key])
#pragma unroll
    for (int c = 0; c < 8; ++c) {
      v16bf bv;
      v8bf u0 = *(const v8bf*)&Vt[c * 16 + lo][hi * 8];
      v8bf u1 = *(const v8bf*)&Vt[c * 16 + lo][16 + hi * 8];
#pragma unroll
      for (int i = 0; i < 8; ++i) { bv[i] = u0[i]; bv[8 + i] = u1[i]; }
      acc[c] = wmma_bf16(ap, bv, acc[c]);
    }
    __syncthreads();   // all waves done with Ks/Vt before next staging
  }

#pragma unroll
  for (int c = 0; c < 8; ++c)
#pragma unroll
    for (int r = 0; r < 8; ++r)
      Yh[((size_t)b * S_ + q0 + hi * 8 + r) * D_ + (size_t)h * DH_ + c * 16 + lo] = acc[c][r];
}

// ---------------------------------------------------------------------------
// Per-(t,h) group norm over DH=128 + gn affine + SiLU gate -> bf16 [T][D].
// One wave32 per (t,h); 4 elements per lane; shuffle reduction.
// ---------------------------------------------------------------------------
__global__ void k_gnorm_gate(const float* __restrict__ Yh, const float* __restrict__ Gf,
                             const float* __restrict__ gns, const float* __restrict__ gnb,
                             __bf16* __restrict__ out) {
  int lane = threadIdx.x & 31, w = threadIdx.x >> 5;
  int item = blockIdx.x * 8 + w;                // t*H + h
  int t = item >> 3, h = item & (H_ - 1);
  const float* y = Yh + (size_t)t * D_ + (size_t)h * DH_;
  float v[4]; float sum = 0.f, sq = 0.f;
#pragma unroll
  for (int j = 0; j < 4; ++j) { v[j] = y[lane * 4 + j]; sum += v[j]; sq += v[j] * v[j]; }
#pragma unroll
  for (int off = 16; off > 0; off >>= 1) {      // wave32 reduction
    sum += __shfl_xor(sum, off, 32);
    sq  += __shfl_xor(sq,  off, 32);
  }
  float mu = sum * (1.f / DH_);
  float var = sq * (1.f / DH_) - mu * mu;
  float inv = rsqrtf(var + 1e-5f);
#pragma unroll
  for (int j = 0; j < 4; ++j) {
    int d = h * DH_ + lane * 4 + j;
    float yn = (v[j] - mu) * inv * gns[d] + gnb[d];
    float g = Gf[(size_t)t * D_ + d];
    float si = g / (1.f + __expf(-g));          // SiLU
    out[(size_t)t * D_ + d] = (__bf16)(si * yn);
  }
}

// ---------------------------------------------------------------------------
// Exact GeLU, fp32 in -> bf16 out
// ---------------------------------------------------------------------------
__global__ void k_gelu(const float* __restrict__ x, __bf16* __restrict__ y, size_t n) {
  size_t i = (size_t)blockIdx.x * 256 + threadIdx.x;
  if (i < n) {
    float v = x[i];
    y[i] = (__bf16)(0.5f * v * (1.f + erff(v * 0.70710678118654752f)));
  }
}

// ---------------------------------------------------------------------------
// Host orchestration
// ---------------------------------------------------------------------------
static inline char* wsalloc(char*& p, size_t bytes) {
  char* r = p;
  p += (bytes + 255) & ~(size_t)255;
  return r;
}

extern "C" void kernel_launch(void* const* d_in, const int* in_sizes, int n_in,
                              void* d_out, int out_size, void* d_ws, size_t ws_size,
                              hipStream_t stream) {
  (void)in_sizes; (void)n_in; (void)out_size; (void)ws_size;
  const int*   ids   = (const int*)d_in[0];
  const float* embed = (const float*)d_in[1];
  const float* proj  = (const float*)d_in[2];
  const float* ln1_s = (const float*)d_in[3];
  const float* ln1_b = (const float*)d_in[4];
  const float* ln2_s = (const float*)d_in[5];
  const float* ln2_b = (const float*)d_in[6];
  const float* Wq    = (const float*)d_in[7];
  const float* Wk    = (const float*)d_in[8];
  const float* Wv    = (const float*)d_in[9];
  const float* Wg    = (const float*)d_in[10];
  const float* Wo    = (const float*)d_in[11];
  const float* gn_s  = (const float*)d_in[12];
  const float* gn_b  = (const float*)d_in[13];
  const float* w1    = (const float*)d_in[14];
  const float* b1    = (const float*)d_in[15];
  const float* w2    = (const float*)d_in[16];
  const float* b2    = (const float*)d_in[17];

  const size_t TD = (size_t)T_ * D_;
  const size_t TF = (size_t)T_ * F_;

  char* p = (char*)d_ws;
  float*  Xf   = (float*) wsalloc(p, TD * 4);
  float*  Yr   = (float*) wsalloc(p, TD * 4);
  float*  Gf   = (float*) wsalloc(p, TD * 4);
  float*  Qf   = (float*) wsalloc(p, TD * 4);
  float*  Kf   = (float*) wsalloc(p, TD * 4);
  float*  Vf   = (float*) wsalloc(p, TD * 4);
  float*  Yhf  = (float*) wsalloc(p, TD * 4);
  float*  H1f  = (float*) wsalloc(p, TF * 4);
  __bf16* Xnb  = (__bf16*)wsalloc(p, TD * 2);
  __bf16* Qb   = (__bf16*)wsalloc(p, TD * 2);
  __bf16* Kb   = (__bf16*)wsalloc(p, TD * 2);
  __bf16* Vb   = (__bf16*)wsalloc(p, TD * 2);
  __bf16* Ygb  = (__bf16*)wsalloc(p, TD * 2);
  __bf16* Znb  = (__bf16*)wsalloc(p, TD * 2);
  __bf16* H1b  = (__bf16*)wsalloc(p, TF * 2);
  __bf16* Xb   = (__bf16*)wsalloc(p, TD * 2);
  __bf16* Wqb  = (__bf16*)wsalloc(p, (size_t)H_ * D_ * DH_ * 2);
  __bf16* Wkb  = (__bf16*)wsalloc(p, (size_t)H_ * D_ * DH_ * 2);
  __bf16* Wvb  = (__bf16*)wsalloc(p, (size_t)H_ * D_ * DH_ * 2);
  __bf16* Wgb  = (__bf16*)wsalloc(p, (size_t)D_ * D_ * 2);
  __bf16* Wob  = (__bf16*)wsalloc(p, (size_t)D_ * D_ * 2);
  __bf16* w1b  = (__bf16*)wsalloc(p, (size_t)D_ * F_ * 2);
  __bf16* w2b  = (__bf16*)wsalloc(p, (size_t)F_ * D_ * 2);
  __bf16* prjb = (__bf16*)wsalloc(p, (size_t)D_ * VOCAB_ * 2);

  auto conv = [&](const float* s, __bf16* d, size_t n) {
    size_t th = (n + 3) / 4;
    k_f32_to_bf16<<<dim3((unsigned)((th + 255) / 256)), 256, 0, stream>>>(s, d, n);
  };
  auto gemm = [&](const __bf16* A, const __bf16* Bm, const float* bias, const float* beta,
                  float* C, int M, int N, int K, int lda, int ldb, int ldc, int flags) {
    dim3 g((unsigned)(N / 128), (unsigned)(M / 64));
    switch (flags) {
      case 0: k_gemm<0><<<g, 256, 0, stream>>>(A, Bm, bias, beta, C, M, N, K, lda, ldb, ldc); break;
      case 1: k_gemm<1><<<g, 256, 0, stream>>>(A, Bm, bias, beta, C, M, N, K, lda, ldb, ldc); break;
      case 2: k_gemm<2><<<g, 256, 0, stream>>>(A, Bm, bias, beta, C, M, N, K, lda, ldb, ldc); break;
      default: k_gemm<3><<<g, 256, 0, stream>>>(A, Bm, bias, beta, C, M, N, K, lda, ldb, ldc); break;
    }
  };

  // X = embed[ids]
  k_embed<<<dim3((unsigned)(TD / 256)), 256, 0, stream>>>(ids, embed, Xf);
  conv(proj, prjb, (size_t)D_ * VOCAB_);

  const size_t WQKV = (size_t)H_ * D_ * DH_;   // per-layer slice of Wq/Wk/Wv

  for (int l = 0; l < L_; ++l) {
    conv(Wq + (size_t)l * WQKV, Wqb, WQKV);
    conv(Wk + (size_t)l * WQKV, Wkb, WQKV);
    conv(Wv + (size_t)l * WQKV, Wvb, WQKV);
    conv(Wg + (size_t)l * D_ * D_, Wgb, (size_t)D_ * D_);
    conv(Wo + (size_t)l * D_ * D_, Wob, (size_t)D_ * D_);
    conv(w1 + (size_t)l * D_ * F_, w1b, (size_t)D_ * F_);
    conv(w2 + (size_t)l * F_ * D_, w2b, (size_t)F_ * D_);

    // Xn = LN1(X)
    k_layernorm<<<dim3(T_), 256, 0, stream>>>(Xf, ln1_s + (size_t)l * D_, ln1_b + (size_t)l * D_, Xnb);

    // per-head QKV projections into [T][H][DH] slabs (ldc = D)
    for (int h = 0; h < H_; ++h) {
      gemm(Xnb, Wqb + (size_t)h * D_ * DH_, nullptr, nullptr, Qf + (size_t)h * DH_,
           T_, DH_, D_, D_, DH_, D_, 0);
      gemm(Xnb, Wkb + (size_t)h * D_ * DH_, nullptr, nullptr, Kf + (size_t)h * DH_,
           T_, DH_, D_, D_, DH_, D_, 0);
      gemm(Xnb, Wvb + (size_t)h * D_ * DH_, nullptr, nullptr, Vf + (size_t)h * DH_,
           T_, DH_, D_, D_, DH_, D_, 0);
    }

    // xPos rotary + bf16 pack
    k_rope_pack<<<dim3((unsigned)((size_t)T_ * H_ * 64 / 256)), 256, 0, stream>>>(Qf, Kf, Vf, Qb, Kb, Vb);

    // fused decayed retention -> Yh  (one block per 128 queries of one (b,h))
    k_retention<<<dim3((unsigned)(B_ * H_ * (S_ / 128))), 256, 0, stream>>>(Qb, Kb, Vb, Yhf);

    // gate G = Xn @ Wg
    gemm(Xnb, Wgb, nullptr, nullptr, Gf, T_, D_, D_, D_, D_, D_, 0);

    // group norm + SiLU gate -> bf16
    k_gnorm_gate<<<dim3((unsigned)(T_ * H_ / 8)), 256, 0, stream>>>(Yhf, Gf,
        gn_s + (size_t)l * D_, gn_b + (size_t)l * D_, Ygb);

    // Yr = gated @ Wo + X
    gemm(Ygb, Wob, nullptr, Xf, Yr, T_, D_, D_, D_, D_, D_, 2);

    // Zn = LN2(Yr)
    k_layernorm<<<dim3(T_), 256, 0, stream>>>(Yr, ln2_s + (size_t)l * D_, ln2_b + (size_t)l * D_, Znb);

    // H1 = Zn @ w1 + b1 ; gelu ; X = H1 @ w2 + b2 + Yr
    gemm(Znb, w1b, b1 + (size_t)l * F_, nullptr, H1f, T_, F_, D_, D_, F_, F_, 1);
    k_gelu<<<dim3((unsigned)((TF + 255) / 256)), 256, 0, stream>>>(H1f, H1b, TF);
    gemm(H1b, w2b, b2 + (size_t)l * D_, Yr, Xf, T_, D_, F_, F_, D_, D_, 3);
  }

  // logits = X @ proj  -> d_out (fp32, T x VOCAB)
  conv(Xf, Xb, TD);
  gemm(Xb, prjb, nullptr, nullptr, (float*)d_out, T_, VOCAB_, D_, D_, VOCAB_, VOCAB_, 0);
}
